// KernelAttentionModule_70360154243344
// MI455X (gfx1250) — matrix operand order repaired and checked by reading, not compiled, not run on hardware
//
#include <hip/hip_runtime.h>
#include <hip/hip_bf16.h>

// Linear-attention module for MI455X (gfx1250, wave32, WMMA).
// Bandwidth-bound (~220MB moved vs ~26 GFLOP): bf16 intermediates, fused
// epilogues, WMMA operands staged in LDS so each fetch is 2x ds_load_b128,
// pure-copy staging via global_load_async_to_lds_b128 (ASYNCcnt path),
// K-step 64 so each stage/barrier amortizes 4-8 WMMAs.

typedef __attribute__((ext_vector_type(16))) __bf16 v16bf;
typedef __attribute__((ext_vector_type(8)))  __bf16 v8bf;
typedef __attribute__((ext_vector_type(4)))  __bf16 v4bf;
typedef __attribute__((ext_vector_type(8)))  float  v8f;

static constexpr int BB = 8;     // batch
static constexpr int CC = 512;   // channels
static constexpr int DD = 64;    // qk dim
static constexpr int NN = 4096;  // tokens (h*w)
#define EPSV 1e-6f

__device__ __forceinline__ int laneid() { return threadIdx.x & 31; }

__device__ __forceinline__ float softplusf(float v) {
    return v > 20.0f ? v : log1pf(__expf(v));
}

// Low 32 bits of a generic LDS pointer are the LDS byte offset (ISA 10.2).
__device__ __forceinline__ unsigned lds_off(const void* p) {
    return (unsigned)(unsigned long long)p;
}

// Async 16B copy global->LDS (CDNA5, tracked by ASYNCcnt). GVS addressing:
// mem = SGPR_base + VGPR_i32 offset. vdst = LDS byte offset.
__device__ __forceinline__ void async_copy16(unsigned ldsaddr, const void* sbase,
                                             unsigned goff) {
    asm volatile("global_load_async_to_lds_b128 %0, %1, %2"
                 :: "v"(ldsaddr), "v"(goff), "s"(sbase) : "memory");
}
__device__ __forceinline__ void async_wait0() {
    asm volatile("s_wait_asynccnt 0" ::: "memory");
}

// ---- WMMA operand gathers (ISA 7.12.2 layouts), 2x ds_load_b128 each ----
// A 16x32 bf16 from row-major LDS tile s[m*ld + k] (ld = 40/72 keeps 16B align)
__device__ __forceinline__ v16bf ldsA_rm(const __bf16* s, int ld) {
    const int lane = laneid();
    const int m = lane & 15;
    const int koff = (lane >> 4) << 3;   // upper half-wave: +8/+24 K-groups
    const __bf16* p = s + m * ld + koff;
    v8bf lo = *(const v8bf*)p;           // k = koff .. koff+7
    v8bf hi = *(const v8bf*)(p + 16);    // k = koff+16 .. koff+23
    return __builtin_shufflevector(lo, hi, 0, 1, 2, 3, 4, 5, 6, 7,
                                   8, 9, 10, 11, 12, 13, 14, 15);
}
// B 32x16 bf16 from column-major LDS tile: B[k][n] = s[n*ld + k].
__device__ __forceinline__ v16bf ldsB_cm(const __bf16* s, int ld) {
    const int lane = laneid();
    const int n = lane & 15;
    const int kb = (lane >> 4) << 3;     // upper half-wave: +8/+24 K-groups
    const __bf16* p = s + n * ld + kb;
    v8bf lo = *(const v8bf*)p;
    v8bf hi = *(const v8bf*)(p + 16);
    return __builtin_shufflevector(lo, hi, 0, 1, 2, 3, 4, 5, 6, 7,
                                   8, 9, 10, 11, 12, 13, 14, 15);
}

#define WMMA_BF16(A, Bm, Cacc) \
    __builtin_amdgcn_wmma_f32_16x16x32_bf16(false, (A), false, (Bm), (short)0, (Cacc), false, false)

// ---------------------------------------------------------------------------
__global__ __launch_bounds__(256) void zero_f32(float* p, int n) {
    int i = blockIdx.x * 256 + threadIdx.x;
    if (i < n) p[i] = 0.0f;
}

// ---------------------------------------------------------------------------
// Kernel 1: fused projection GEMM  [wq;wk;wv](640x512) x xf(512x4096) / batch.
// 64-row blocks are homogeneous: rb=0 -> wq(+softplus), rb=1 -> wk(+softplus),
// rb>=2 -> wv rows. All source/dest selection is block-uniform (scalar).
__global__ __launch_bounds__(256)
void proj_kernel(const float* __restrict__ x,
                 const float* __restrict__ wq, const float* __restrict__ bq,
                 const float* __restrict__ wk, const float* __restrict__ bk,
                 const float* __restrict__ wv, const float* __restrict__ bv,
                 __bf16* __restrict__ Qs, __bf16* __restrict__ Ks,
                 __bf16* __restrict__ Vs) {
    __shared__ __attribute__((aligned(16))) __bf16 Ws[64 * 72];  // A: [row][k]
    __shared__ __attribute__((aligned(16))) __bf16 Xs[64 * 72];  // B: [tok][k]
    __shared__ float Bs[64];
    const int rb = blockIdx.x;          // 0..9
    const int n0 = blockIdx.y * 64;     // token base
    const int bz = blockIdx.z;
    const int tid = threadIdx.x;
    const int w = tid >> 5, lane = tid & 31;
    const int mt = w & 3, nt0 = (w >> 2) << 1;

    // Block-uniform source select (scalar branch, hoisted out of all loops).
    const float* wbase;
    const float* bbase;
    if (rb == 0)      { wbase = wq; bbase = bq; }
    else if (rb == 1) { wbase = wk; bbase = bk; }
    else              { wbase = wv + (size_t)(rb - 2) * 64 * CC;
                        bbase = bv + (rb - 2) * 64; }
    if (tid < 64) Bs[tid] = bbase[tid];

    v8f acc[2] = {};

    for (int kk = 0; kk < CC; kk += 64) {
        // stage weights 64x64 (row-major): float4 read, v4bf (b64) LDS write
#pragma unroll
        for (int i = 0; i < 4; ++i) {
            int idx = tid + 256 * i;
            int r = idx >> 4, k = (idx & 15) * 4;
            float4 f = *(const float4*)(wbase + (size_t)r * CC + kk + k);
            v4bf h = { (__bf16)f.x, (__bf16)f.y, (__bf16)f.z, (__bf16)f.w };
            *(v4bf*)(Ws + r * 72 + k) = h;
        }
        // stage x 64x64 -> column-major [tok][k]: float4 along tokens
#pragma unroll
        for (int i = 0; i < 4; ++i) {
            int idx = tid + 256 * i;
            int k = idx >> 4, n = (idx & 15) * 4;
            float4 f = *(const float4*)(x + ((size_t)bz * CC + kk + k) * NN + n0 + n);
            Xs[(n + 0) * 72 + k] = (__bf16)f.x;
            Xs[(n + 1) * 72 + k] = (__bf16)f.y;
            Xs[(n + 2) * 72 + k] = (__bf16)f.z;
            Xs[(n + 3) * 72 + k] = (__bf16)f.w;
        }
        __syncthreads();
        v16bf a0 = ldsA_rm(Ws + mt * 16 * 72, 72);
        v16bf a1 = ldsA_rm(Ws + mt * 16 * 72 + 32, 72);
#pragma unroll
        for (int t = 0; t < 2; ++t) {
            v16bf b0 = ldsB_cm(Xs + (nt0 + t) * 16 * 72, 72);
            v16bf b1 = ldsB_cm(Xs + (nt0 + t) * 16 * 72 + 32, 72);
            acc[t] = WMMA_BF16(a0, b0, acc[t]);
            acc[t] = WMMA_BF16(a1, b1, acc[t]);
        }
        __syncthreads();
    }

    const int coln = lane & 15, rowadd = (lane >> 4) << 3;
    if (rb <= 1) {  // Q or K rows: softplus
        __bf16* dst = (rb == 0 ? Qs : Ks) + (size_t)bz * DD * NN;
#pragma unroll
        for (int t = 0; t < 2; ++t) {
#pragma unroll
            for (int r = 0; r < 8; ++r) {
                int row_l = mt * 16 + r + rowadd;
                int n = n0 + (nt0 + t) * 16 + coln;
                dst[(size_t)row_l * NN + n] = (__bf16)softplusf(acc[t][r] + Bs[row_l]);
            }
        }
    } else {        // V rows
        __bf16* dst = Vs + ((size_t)bz * CC + (rb - 2) * 64) * NN;
#pragma unroll
        for (int t = 0; t < 2; ++t) {
#pragma unroll
            for (int r = 0; r < 8; ++r) {
                int row_l = mt * 16 + r + rowadd;
                int n = n0 + (nt0 + t) * 16 + coln;
                dst[(size_t)row_l * NN + n] = (__bf16)(acc[t][r] + Bs[row_l]);
            }
        }
    }
}

// ---------------------------------------------------------------------------
// Kernel 2: Ksum[b,d] = sum_n K[b,d,n] + EPS
__global__ __launch_bounds__(256)
void ksum_kernel(const __bf16* __restrict__ Ks, float* __restrict__ Ksum) {
    const int bd = blockIdx.x;  // 0..511
    const v8bf* p = (const v8bf*)(Ks + (size_t)bd * NN);
    float s = 0.0f;
    for (int i = threadIdx.x; i < NN / 8; i += 256) {
        v8bf v = p[i];
#pragma unroll
        for (int j = 0; j < 8; ++j) s += (float)v[j];
    }
#pragma unroll
    for (int off = 16; off > 0; off >>= 1) s += __shfl_down(s, off, 32);
    __shared__ float red[8];
    if ((threadIdx.x & 31) == 0) red[threadIdx.x >> 5] = s;
    __syncthreads();
    if (threadIdx.x == 0) {
        float t = 0.0f;
#pragma unroll
        for (int i = 0; i < 8; ++i) t += red[i];
        Ksum[bd] = t + EPSV;
    }
}

// ---------------------------------------------------------------------------
// Kernel 3: KV[b,m,c] += K(64 x n-chunk) x V^T. Pure-copy staging via async
// global->LDS B128 copies (ASYNCcnt). Split-K over n, f32 atomics. K-step 64.
__global__ __launch_bounds__(256)
void kv_kernel(const __bf16* __restrict__ Ks, const __bf16* __restrict__ Vs,
               float* __restrict__ KV) {
    __shared__ __attribute__((aligned(16))) __bf16 Kt[64 * 72];   // A: [m][kn]
    __shared__ __attribute__((aligned(16))) __bf16 Vt[128 * 72];  // B: [c][kn]
    const int cb = blockIdx.x;   // c base cb*128
    const int nb = blockIdx.y;   // n chunk of 512
    const int bz = blockIdx.z;
    const int tid = threadIdx.x, w = tid >> 5, lane = tid & 31;
    const int mt = w & 3, ct0 = (w >> 2) << 2;
    const unsigned kt0 = lds_off(Kt), vt0 = lds_off(Vt);
    v8f acc[4] = {};

    for (int kk = 0; kk < 512; kk += 64) {
        const int n0 = nb * 512 + kk;
#pragma unroll
        for (int i = 0; i < 2; ++i) {  // K tile 64x64: 512 chunks of 8 bf16
            int idx = tid + 256 * i;
            int m = idx >> 3, kq = (idx & 7) * 8;
            async_copy16(kt0 + (unsigned)((m * 72 + kq) * 2), Ks,
                         (unsigned)((((bz * DD + m) * NN) + n0 + kq) * 2));
        }
#pragma unroll
        for (int i = 0; i < 4; ++i) {  // V tile 128x64: 1024 chunks
            int idx = tid + 256 * i;
            int c = idx >> 3, kq = (idx & 7) * 8;
            async_copy16(vt0 + (unsigned)((c * 72 + kq) * 2), Vs,
                         (unsigned)((((bz * CC + cb * 128 + c) * NN) + n0 + kq) * 2));
        }
        async_wait0();
        __syncthreads();
        v16bf a0 = ldsA_rm(Kt + mt * 16 * 72, 72);
        v16bf a1 = ldsA_rm(Kt + mt * 16 * 72 + 32, 72);
#pragma unroll
        for (int j = 0; j < 4; ++j) {
            v16bf b0 = ldsB_cm(Vt + (ct0 + j) * 16 * 72, 72);
            v16bf b1 = ldsB_cm(Vt + (ct0 + j) * 16 * 72 + 32, 72);
            acc[j] = WMMA_BF16(a0, b0, acc[j]);
            acc[j] = WMMA_BF16(a1, b1, acc[j]);
        }
        __syncthreads();
    }
    const int coln = lane & 15, rowadd = (lane >> 4) << 3;
#pragma unroll
    for (int j = 0; j < 4; ++j) {
#pragma unroll
        for (int r = 0; r < 8; ++r) {
            int m = mt * 16 + r + rowadd;
            int c = cb * 128 + (ct0 + j) * 16 + coln;
            __hip_atomic_fetch_add(&KV[((size_t)bz * DD + m) * CC + c], acc[j][r],
                                   __ATOMIC_RELAXED, __HIP_MEMORY_SCOPE_AGENT);
        }
    }
}

// ---------------------------------------------------------------------------
// Kernel 4: out[b,c,n] = x + gamma*norm[n]*sum_m KV[b,m,c]*Q[b,m,n],
// norm[n] = 1/sum_m Q[b,m,n]*Ksum[b,m]. Tile 64c x 64tok, inner m=64.
__global__ __launch_bounds__(256)
void out_kernel(const float* __restrict__ x, const float* __restrict__ KV,
                const float* __restrict__ Ksum, const __bf16* __restrict__ Qs,
                const float* __restrict__ gptr, float* __restrict__ out) {
    __shared__ __attribute__((aligned(16))) __bf16 KVc[64 * 72];  // A: [c][m]
    __shared__ __attribute__((aligned(16))) __bf16 Qc[64 * 72];   // B: [tok][m]
    __shared__ float KsL[64];
    __shared__ float normL[64];
    const int n0 = blockIdx.x * 64;
    const int c0 = blockIdx.y * 64;
    const int bz = blockIdx.z;
    const int tid = threadIdx.x, w = tid >> 5, lane = tid & 31;

    // stage KV (f32->bf16, transpose to [c][m]) and Q (transpose to [tok][m])
#pragma unroll
    for (int i = 0; i < 4; ++i) {
        int idx = tid + 256 * i;
        int m = idx >> 4, c = (idx & 15) * 4;
        float4 f = *(const float4*)(KV + ((size_t)bz * DD + m) * CC + c0 + c);
        KVc[(c + 0) * 72 + m] = (__bf16)f.x;
        KVc[(c + 1) * 72 + m] = (__bf16)f.y;
        KVc[(c + 2) * 72 + m] = (__bf16)f.z;
        KVc[(c + 3) * 72 + m] = (__bf16)f.w;
    }
#pragma unroll
    for (int i = 0; i < 2; ++i) {
        int idx = tid + 256 * i;
        int m = idx >> 3, t = (idx & 7) * 8;
        v8bf q = *(const v8bf*)(Qs + ((size_t)bz * DD + m) * NN + n0 + t);
#pragma unroll
        for (int j = 0; j < 8; ++j) Qc[(t + j) * 72 + m] = q[j];
    }
    if (tid < 64) KsL[tid] = Ksum[bz * DD + tid];
    __syncthreads();
    if (tid < 64) {  // per-token normalizer (contiguous row of Qc)
        float s = 0.0f;
#pragma unroll
        for (int m = 0; m < 64; ++m) s += (float)Qc[tid * 72 + m] * KsL[m];
        normL[tid] = 1.0f / s;
    }
    __syncthreads();

    const int ct = w & 3, tt0 = (w >> 2) << 1;
    v8f acc[2] = {};
#pragma unroll
    for (int ks = 0; ks < 2; ++ks) {                  // m = 0..31, 32..63
        v16bf a = ldsA_rm(KVc + ct * 16 * 72 + ks * 32, 72);
#pragma unroll
        for (int t = 0; t < 2; ++t) {
            v16bf b = ldsB_cm(Qc + (tt0 + t) * 16 * 72 + ks * 32, 72);
            acc[t] = WMMA_BF16(a, b, acc[t]);
        }
    }

    const float g = gptr[0];
    const int coln = lane & 15, rowadd = (lane >> 4) << 3;
#pragma unroll
    for (int t = 0; t < 2; ++t) {
        int tl = (tt0 + t) * 16 + coln;
        float scale = normL[tl] * g;
        int n = n0 + tl;
#pragma unroll
        for (int r = 0; r < 8; ++r) {
            int c = c0 + ct * 16 + r + rowadd;
            size_t off = ((size_t)bz * CC + c) * NN + n;
            out[off] = x[off] + scale * acc[t][r];
        }
    }
}

// ---------------------------------------------------------------------------
extern "C" void kernel_launch(void* const* d_in, const int* in_sizes, int n_in,
                              void* d_out, int out_size, void* d_ws, size_t ws_size,
                              hipStream_t stream) {
    (void)in_sizes; (void)n_in; (void)out_size; (void)ws_size;
    const float* x  = (const float*)d_in[0];
    const float* wq = (const float*)d_in[1];
    const float* bq = (const float*)d_in[2];
    const float* wk = (const float*)d_in[3];
    const float* bk = (const float*)d_in[4];
    const float* wv = (const float*)d_in[5];
    const float* bv = (const float*)d_in[6];
    const float* gm = (const float*)d_in[7];
    float* out = (float*)d_out;

    // Workspace (~43 MB): Q bf16 | K bf16 | V bf16 | KV f32 | Ksum f32
    char* ws = (char*)d_ws;
    __bf16* Qs = (__bf16*)ws;
    __bf16* Ks = Qs + (size_t)BB * DD * NN;
    __bf16* Vs = Ks + (size_t)BB * DD * NN;
    float*  KV = (float*)(Vs + (size_t)BB * CC * NN);
    float*  Ksum = KV + (size_t)BB * DD * CC;

    zero_f32<<<dim3((BB * DD * CC + 255) / 256), 256, 0, stream>>>(KV, BB * DD * CC);
    proj_kernel<<<dim3(10, NN / 64, BB), 256, 0, stream>>>(x, wq, bq, wk, bk, wv, bv,
                                                           Qs, Ks, Vs);
    ksum_kernel<<<dim3(BB * DD), 256, 0, stream>>>(Ks, Ksum);
    kv_kernel<<<dim3(CC / 128, NN / 512, BB), 256, 0, stream>>>(Ks, Vs, KV);
    out_kernel<<<dim3(NN / 64, CC / 64, BB), 256, 0, stream>>>(x, KV, Ksum, Qs, gm, out);
}
